// SpeakerEncoder_59296318488879
// MI455X (gfx1250) — compile-verified, matrix operand order checked by reading, not run on hardware
//
#include <hip/hip_runtime.h>
#include <stdint.h>

#define S_SPK 1024
#define U_UTT 64
#define E_DIM 256
#define SU    (S_SPK * U_UTT)

#define ROWS  128          // rows (utterances) per block in GEMM kernel
#define CHUNK 128          // columns (speakers) per LDS chunk
#define KP    264          // padded bf16 K-stride: 264*2=528 B, 16B-aligned

typedef __attribute__((ext_vector_type(16))) __bf16 v16bf;
typedef __attribute__((ext_vector_type(8)))  float  v8f;

union Frag16 { v16bf bf; uint4 u4[2]; };

__device__ __forceinline__ unsigned short f2bf(float f) {
  unsigned u = __float_as_uint(f);
  unsigned r = u + 0x7FFFu + ((u >> 16) & 1u);   // round-to-nearest-even
  return (unsigned short)(r >> 16);
}

// ---------------------------------------------------------------------------
// Kernel 0: zero the output accumulators (harness poisons d_out each run)
// ---------------------------------------------------------------------------
__global__ void ge2e_zero(float* out) {
  if (threadIdx.x < 2) out[threadIdx.x] = 0.f;
}

// ---------------------------------------------------------------------------
// Kernel 1: per-speaker centroids.
//  - B matrix (c_incl) as bf16, COLUMN-major [s][k=256] so a GEMM column
//    chunk is one contiguous 64 KB region (async-copy friendly)
//  - invn[s]   = 1 / ||c_incl_s||
//  - diag[s,u] = w * cos(e_su, c_excl_su) + b   (closed form, U-1 cancels)
// ---------------------------------------------------------------------------
__global__ __launch_bounds__(256)
void ge2e_centroids(const float* __restrict__ e,
                    const float* __restrict__ simw,
                    const float* __restrict__ simb,
                    unsigned short* __restrict__ Bb,
                    float* __restrict__ invn,
                    float* __restrict__ diag)
{
  __shared__ float colsum[E_DIM];
  __shared__ float red[8];
  const int s    = blockIdx.x;
  const int t    = threadIdx.x;
  const int lane = t & 31;
  const int wid  = t >> 5;
  const float* es = e + (size_t)s * U_UTT * E_DIM;

  float sum = 0.f;
  #pragma unroll 8
  for (int u = 0; u < U_UTT; ++u) sum += es[u * E_DIM + t];
  colsum[t] = sum;
  Bb[s * E_DIM + t] = f2bf(sum * (1.f / (float)U_UTT));   // col-major c_incl

  float x = sum * sum;                                    // ||colsum||^2
  #pragma unroll
  for (int m = 16; m >= 1; m >>= 1) x += __shfl_xor(x, m, 32);
  if (lane == 0) red[wid] = x;
  __syncthreads();
  float cs2 = 0.f;
  #pragma unroll
  for (int i = 0; i < 8; ++i) cs2 += red[i];
  if (t == 0) invn[s] = (float)U_UTT * rsqrtf(cs2);       // 1/||colsum/64||

  const float w = simw[0], b = simb[0];
  for (int r = 0; r < 8; ++r) {
    const int u = wid * 8 + r;
    const float* eu = es + u * E_DIM;
    float dot = 0.f, ne2 = 0.f;
    #pragma unroll
    for (int i = 0; i < 8; ++i) {
      const int d = lane + 32 * i;
      const float v = eu[d];                              // L2-resident re-read
      dot += v * colsum[d];
      ne2 += v * v;
    }
    #pragma unroll
    for (int m = 16; m >= 1; m >>= 1) {
      dot += __shfl_xor(dot, m, 32);
      ne2 += __shfl_xor(ne2, m, 32);
    }
    if (lane == 0) {
      const float num  = dot - ne2;
      const float den2 = cs2 - 2.f * dot + ne2;
      const float c    = num * rsqrtf(fmaxf(den2, 1e-24f));
      diag[s * U_UTT + u] = w * c + b;
    }
  }
}

// ---------------------------------------------------------------------------
// Kernel 2: fused bf16-WMMA GEMM + online log-softmax + CE loss + accuracy.
// Block = 128 rows x all 1024 cols (8 chunks of 128). 8 waves, 16 rows/wave.
// B chunks staged via GLOBAL_LOAD_ASYNC_TO_LDS_B128 (ASYNCcnt-tracked).
// ---------------------------------------------------------------------------
__global__ __launch_bounds__(256)
void ge2e_sim_loss(const float* __restrict__ e,
                   const unsigned short* __restrict__ Bb,
                   const float* __restrict__ invn,
                   const float* __restrict__ diag,
                   const float* __restrict__ simw,
                   const float* __restrict__ simb,
                   float* __restrict__ out)
{
  __shared__ unsigned short smA[ROWS * KP];    // 66 KB, row-major [row][k]
  __shared__ unsigned short smB[CHUNK * KP];   // 66 KB, col-major [col][k]
  __shared__ float sInvn[S_SPK];               // 4 KB

  const int tid  = threadIdx.x;
  const int lane = tid & 31;
  const int wid  = tid >> 5;
  const int half = lane >> 4;
  const int l16  = lane & 15;
  const int rowBase = blockIdx.x * ROWS;

  #pragma unroll
  for (int i = 0; i < S_SPK / 256; ++i)
    sInvn[tid + 256 * i] = invn[tid + 256 * i];

  // Stage A tile: ROWS x 256 fp32 -> bf16 LDS (8192 float4, 32 per thread)
  for (int i = 0; i < 32; ++i) {
    const int idx = tid + 256 * i;
    const int r = idx >> 6, kq = idx & 63;
    const float4 v = *(const float4*)(e + (size_t)(rowBase + r) * E_DIM + kq * 4);
    uint2 p;
    p.x = (unsigned)f2bf(v.x) | ((unsigned)f2bf(v.y) << 16);
    p.y = (unsigned)f2bf(v.z) | ((unsigned)f2bf(v.w) << 16);
    *(uint2*)(smA + r * KP + kq * 4) = p;
  }

  const float wgt = simw[0], bia = simb[0];
  const int j = (rowBase + wid * 16) >> 6;     // speaker id for this wave's rows

  float dlog[8];                               // diagonal logits, VGPR i -> row
  {
    const int r0 = rowBase + wid * 16 + half * 8;
    #pragma unroll
    for (int i = 0; i < 8; ++i) dlog[i] = diag[r0 + i];
  }

  float mrow[8], srow[8]; int crow[8];
  #pragma unroll
  for (int i = 0; i < 8; ++i) { mrow[i] = -3.4e38f; srow[i] = 0.f; crow[i] = -1; }

  const unsigned smBbase = (unsigned)(uintptr_t)&smB[0];   // LDS byte offset

  for (int chunk = 0; chunk < S_SPK / CHUNK; ++chunk) {
    __syncthreads();   // previous chunk's fragment reads complete
    // --- async-stage B chunk: contiguous 64 KB global -> padded col-major LDS
    {
      const unsigned long long gbase =
          (unsigned long long)(uintptr_t)(Bb + (size_t)chunk * CHUNK * E_DIM);
      #pragma unroll
      for (int i = 0; i < 16; ++i) {
        const int idx = tid + 256 * i;           // 16-byte granules
        const int col = idx >> 5, kq = idx & 31; // 32 granules per column
        const unsigned      ldsa = smBbase + (unsigned)(col * (KP * 2) + kq * 16);
        const unsigned long long ga = gbase + (unsigned long long)idx * 16ull;
        asm volatile("global_load_async_to_lds_b128 %0, %1, off"
                     :: "v"(ldsa), "v"(ga) : "memory");
      }
      asm volatile("s_wait_asynccnt 0" ::: "memory");
    }
    __syncthreads();

    v8f acc[8];
    #pragma unroll
    for (int t = 0; t < 8; ++t) acc[t] = (v8f){0.f,0.f,0.f,0.f,0.f,0.f,0.f,0.f};

    #pragma unroll 1
    for (int ks = 0; ks < 8; ++ks) {
      const int kb = ks * 32 + half * 8;       // per-lane K base (bf16 A/B layout)
      Frag16 a;
      {
        const unsigned short* pa = smA + (wid * 16 + l16) * KP + kb;
        a.u4[0] = *(const uint4*)(pa);
        a.u4[1] = *(const uint4*)(pa + 16);
      }
      #pragma unroll
      for (int t = 0; t < 8; ++t) {
        Frag16 bfg;
        const unsigned short* pb = smB + (t * 16 + l16) * KP + kb;
        bfg.u4[0] = *(const uint4*)(pb);
        bfg.u4[1] = *(const uint4*)(pb + 16);
        acc[t] = __builtin_amdgcn_wmma_f32_16x16x32_bf16(
            false, a.bf, false, bfg.bf, (short)0, acc[t], false, false);
      }
    }

    // Epilogue: scale, diagonal substitution, branchless online softmax
    const bool dchunk = (chunk == (j >> 7));
    const int  tgt    = (j >> 4) & 7;
    const bool lhit   = (l16 == (j & 15));
    #pragma unroll
    for (int t = 0; t < 8; ++t) {
      const int col = chunk * CHUNK + t * 16 + l16;
      const float sc = sInvn[col] * wgt;
      const bool dsub = dchunk && lhit && (t == tgt);
      #pragma unroll
      for (int i = 0; i < 8; ++i) {
        float v = acc[t][i] * sc + bia;
        if (dsub) v = dlog[i];
        const float nm = fmaxf(mrow[i], v);
        srow[i] = srow[i] * __expf(mrow[i] - nm) + __expf(v - nm);
        crow[i] = (v > mrow[i]) ? col : crow[i];
        mrow[i] = nm;
      }
    }
  }

  // Merge (m, sumexp, argmax) across the 16 lanes sharing each row
  #pragma unroll
  for (int i = 0; i < 8; ++i) {
    float m = mrow[i], s = srow[i]; int c = crow[i];
    #pragma unroll
    for (int msk = 8; msk >= 1; msk >>= 1) {
      const float om = __shfl_xor(m, msk, 32);
      const float os = __shfl_xor(s, msk, 32);
      const int   oc = __shfl_xor(c, msk, 32);
      const float nm = fmaxf(m, om);
      s = s * __expf(m - nm) + os * __expf(om - nm);
      c = (om > m) ? oc : c;
      m = nm;
    }
    mrow[i] = m; srow[i] = s; crow[i] = c;
  }

  if (l16 == 0) {
    float lsum = 0.f, asum = 0.f;
    #pragma unroll
    for (int i = 0; i < 8; ++i) {
      lsum += (mrow[i] + logf(srow[i])) - dlog[i];   // -log p(gt); gt col is diag
      asum += (crow[i] == j) ? 1.f : 0.f;
    }
    const float inv = 1.f / (float)SU;
    atomicAdd(&out[0], lsum * inv);
    atomicAdd(&out[1], asum * inv);
  }
}

// ---------------------------------------------------------------------------
extern "C" void kernel_launch(void* const* d_in, const int* in_sizes, int n_in,
                              void* d_out, int out_size, void* d_ws, size_t ws_size,
                              hipStream_t stream) {
  const float* e    = (const float*)d_in[0];
  const float* simw = (const float*)d_in[1];
  const float* simb = (const float*)d_in[2];
  float* out = (float*)d_out;

  unsigned short* Bb  = (unsigned short*)d_ws;                       // 512 KB
  float* invn = (float*)((char*)d_ws + 512 * 1024);                  // 4 KB
  float* diag = (float*)((char*)d_ws + 512 * 1024 + 4096);           // 256 KB

  ge2e_zero<<<1, 32, 0, stream>>>(out);
  ge2e_centroids<<<S_SPK, 256, 0, stream>>>(e, simw, simb, Bb, invn, diag);
  ge2e_sim_loss<<<SU / ROWS, 256, 0, stream>>>(e, Bb, invn, diag, simw, simb, out);
}
